// MPNNNet_45097156608288
// MI455X (gfx1250) — matrix operand order repaired and checked by reading, not compiled, not run on hardware
//
#include <hip/hip_runtime.h>
#include <hip/hip_bf16.h>
#include <math.h>

// ---------------------------------------------------------------------------
// Types for CDNA5 WMMA (wave32) + TDM descriptors
// ---------------------------------------------------------------------------
typedef __attribute__((ext_vector_type(16))) __bf16 bf16x16;
typedef __attribute__((ext_vector_type(8)))  __bf16 bf16x8;
typedef __attribute__((ext_vector_type(4)))  __bf16 bf16x4;
typedef __attribute__((ext_vector_type(8)))  float  floatx8;
typedef __attribute__((ext_vector_type(4)))  unsigned int uint32x4;
typedef __attribute__((ext_vector_type(8)))  int    int32x8;
typedef __attribute__((ext_vector_type(4)))  int    int32x4;

#if defined(__HIP_DEVICE_COMPILE__) && __has_builtin(__builtin_amdgcn_tensor_load_to_lds)
  #define HAVE_TDM 1
#else
  #define HAVE_TDM 0
#endif

// ---------------------------------------------------------------------------
// Specialized tiled GEMM: C[M,N] = act(A[M,K] @ B[K,N] + bias), bf16 WMMA.
// Compile-time K / NCOLS so the aligned shapes are guard-free and the K loop
// fully unrolls into back-to-back v_wmma_f32_16x16x32_bf16.
// Requirements used here: gridDim.x*16 == M, N % NCOLS == 0.
// ---------------------------------------------------------------------------
template<int K, int NCOLS, bool RELU, bool OUT_BF16>
__global__ __launch_bounds__(NCOLS * 2)
void gemm_wmma(const float* __restrict__ A, const float* __restrict__ B,
               const float* __restrict__ bias, void* __restrict__ C, int N)
{
    constexpr int THREADS = NCOLS * 2;         // 1 wave per 16-col tile
    constexpr int NCH     = (K + 31) / 32;     // K chunks of 32
    constexpr bool A_VEC  = (K % 4 == 0);      // float4-loadable A rows

    const int tid  = threadIdx.x;
    const int wave = tid >> 5;
    const int lane = tid & 31;
    const int half = lane >> 4;                // 0: lanes 0-15, 1: lanes 16-31
    const int r    = lane & 15;

    const int m0  = blockIdx.x * 16;
    const int bn0 = blockIdx.y * NCOLS;

    __shared__ __bf16 lA [16][32];             // A tile, row-major [m][k]
    __shared__ __bf16 lBt[NCOLS][32];          // B tile, transposed [n][k]

    floatx8 acc = {};

    #pragma unroll
    for (int c = 0; c < NCH; ++c) {
        const int k0 = c * 32;

        // ---------------- stage A tile (16 x 32) ----------------
        if (A_VEC && (k0 + 32 <= K)) {
            #pragma unroll
            for (int idx = tid; idx < 128; idx += THREADS) {   // 128 float4s
                int mm = idx >> 3, q = idx & 7;
                float4 v = *(const float4*)&A[(size_t)(m0 + mm) * K + k0 + 4 * q];
                bf16x4 p = { (__bf16)v.x, (__bf16)v.y, (__bf16)v.z, (__bf16)v.w };
                *(bf16x4*)&lA[mm][4 * q] = p;
            }
        } else {
            for (int idx = tid; idx < 512; idx += THREADS) {
                int mm = idx >> 5, kk = idx & 31;
                int gk = k0 + kk;
                float v = (gk < K) ? A[(size_t)(m0 + mm) * K + gk] : 0.0f;
                lA[mm][kk] = (__bf16)v;
            }
        }

        // ---------------- stage B tile (32 x NCOLS, transposed) ----------------
        if (k0 + 32 <= K) {
            constexpr int QC = NCOLS / 4;                      // float4s per row
            #pragma unroll
            for (int it = 0; it < (32 * QC) / THREADS; ++it) { // == 4
                int idx = tid + it * THREADS;
                int kk  = idx / QC;
                int nn  = (idx - kk * QC) * 4;
                float4 v = *(const float4*)&B[(size_t)(k0 + kk) * N + bn0 + nn];
                lBt[nn + 0][kk] = (__bf16)v.x;
                lBt[nn + 1][kk] = (__bf16)v.y;
                lBt[nn + 2][kk] = (__bf16)v.z;
                lBt[nn + 3][kk] = (__bf16)v.w;
            }
        } else {
            for (int idx = tid; idx < 32 * NCOLS; idx += THREADS) {
                int kk = idx / NCOLS;
                int nn = idx - kk * NCOLS;
                int gk = k0 + kk;
                float v = (gk < K) ? B[(size_t)gk * N + bn0 + nn] : 0.0f;
                lBt[nn][kk] = (__bf16)v;
            }
        }
        __syncthreads();

        // ---- fragments per ISA 7.12.2 (16-bit A 16x32 / B 32x16, wave32) ----
        // A lane(0-15): M=r, K = {half*8..+7} U {16+half*8..+7}  (contiguous runs)
        bf16x8 alo = *(const bf16x8*)&lA[r][half * 8];
        bf16x8 ahi = *(const bf16x8*)&lA[r][16 + half * 8];
        bf16x16 afrag = __builtin_shufflevector(alo, ahi,
                            0,1,2,3,4,5,6,7,8,9,10,11,12,13,14,15);
        // B lane(0-15): N=col, K = {half*8..+7} U {16+half*8..+7}  (contiguous runs)
        const int ncol = wave * 16 + r;
        bf16x8 blo = *(const bf16x8*)&lBt[ncol][half * 8];
        bf16x8 bhi = *(const bf16x8*)&lBt[ncol][16 + half * 8];
        bf16x16 bfrag = __builtin_shufflevector(blo, bhi,
                            0,1,2,3,4,5,6,7,8,9,10,11,12,13,14,15);

        acc = __builtin_amdgcn_wmma_f32_16x16x32_bf16(
                  false, afrag, false, bfrag, (short)0, acc, false, false);
        __syncthreads();
    }

    // ---- store: C 16x16 f32 layout: vgpr j -> row j + 8*half, col = r ----
    const int gn = bn0 + wave * 16 + r;
    const float bv = bias[gn];
    #pragma unroll
    for (int j = 0; j < 8; ++j) {
        const int gm = m0 + half * 8 + j;
        float v = acc[j] + bv;
        if (RELU) v = fmaxf(v, 0.0f);
        if (OUT_BF16) ((__bf16*)C)[(size_t)gm * N + gn] = (__bf16)v;
        else          ((float*)C)[(size_t)gm * N + gn] = v;
    }
}

// ---------------------------------------------------------------------------
// Message + scatter. One wave per edge: DMA the edge's 8KB bf16 We tile into
// this wave's LDS slice with the Tensor Data Mover, s_wait_tensorcnt, then a
// 64x64 mat-vec from LDS (2 outputs per lane), atomic scatter to dst.
// ---------------------------------------------------------------------------
#define MSG_WAVES 8   // waves (= edges) per block; E % MSG_WAVES == 0

__global__ __launch_bounds__(MSG_WAVES * 32)
void msg_scatter(const float* __restrict__ x,
                 const __bf16* __restrict__ We,
                 const int* __restrict__ src,
                 const int* __restrict__ dst,
                 float* __restrict__ agg, int E)
{
    __shared__ __bf16 tile[MSG_WAVES][4096];

    const int wave = threadIdx.x >> 5;
    const int lane = threadIdx.x & 31;
    const int e    = blockIdx.x * MSG_WAVES + wave;
    if (e >= E) return;

#if HAVE_TDM
    // ---- build Tensor DMA Descriptor (ISA 8.3/8.4): 1D-ish 4096-elem tile ----
    const unsigned long long ga =
        (unsigned long long)(const void*)(We + (size_t)e * 4096);
    const unsigned int lds = (unsigned int)(uintptr_t)&tile[wave][0];

    uint32x4 g0;
    g0[0] = 1u;                                   // count=1, user descriptor
    g0[1] = lds;                                  // lds_addr (bytes)
    g0[2] = (unsigned int)ga;                     // global_addr[31:0]
    g0[3] = ((unsigned int)(ga >> 32) & 0x1FFFFFFu) | (2u << 30); // addr[56:32] | type=2

    const unsigned int td0 = 4096, tile0 = 4096, str0 = 4096;
    int32x8 g1;
    g1[0] = (int)(1u << 16);                      // data_size = 2 bytes
    g1[1] = (int)((td0 & 0xFFFFu) << 16);         // tensor_dim0[15:0] @ bits 63:48
    g1[2] = (int)((td0 >> 16) | (1u << 16));      // tensor_dim0[31:16] | tensor_dim1=1
    g1[3] = (int)((0u) | (tile0 << 16));          // tensor_dim1[31:16]=0 | tile_dim0
    g1[4] = (int)(1u);                            // tile_dim1=1, tile_dim2=0
    g1[5] = (int)str0;                            // tensor_dim0_stride[31:0]
    g1[6] = (int)((0u) | ((str0 & 0xFFFFu) << 16)); // stride0[47:32]=0 | stride1[15:0]
    g1[7] = (int)(str0 >> 16);                    // stride1[47:16]

    const int32x4 gz = {};
  #if defined(__clang_major__) && (__clang_major__ >= 23)
    const int32x8 gz8 = {};
    __builtin_amdgcn_tensor_load_to_lds(g0, g1, gz, gz, gz8, 0);
  #else
    __builtin_amdgcn_tensor_load_to_lds(g0, g1, gz, gz, 0);
  #endif
    __builtin_amdgcn_s_wait_tensorcnt(0);

    const __bf16* W = &tile[wave][0];
#else
    const __bf16* W = We + (size_t)e * 4096;      // fallback: direct global (L2)
#endif

    const int s = src[e], d = dst[e];
    const float* __restrict__ xs = x + (size_t)s * 64;

    float a0 = 0.0f, a1 = 0.0f;
    #pragma unroll 8
    for (int i = 0; i < 64; ++i) {
        const float xv = xs[i];
        a0 += xv * (float)W[i * 64 + lane];
        a1 += xv * (float)W[i * 64 + lane + 32];
    }
    atomicAdd(&agg[(size_t)d * 64 + lane],      a0);
    atomicAdd(&agg[(size_t)d * 64 + lane + 32], a1);
}

// ---------------------------------------------------------------------------
// Small elementwise / reduction kernels
// ---------------------------------------------------------------------------
__global__ void zero_f32(float* __restrict__ p, long n)
{
    long i = (long)blockIdx.x * blockDim.x + threadIdx.x;
    if (i < n) p[i] = 0.0f;
}

__global__ void bias_relu64(const float* __restrict__ agg,
                            const float* __restrict__ b,
                            float* __restrict__ m, int V)
{
    int idx = blockIdx.x * blockDim.x + threadIdx.x;
    if (idx >= V * 64) return;
    m[idx] = fmaxf(agg[idx] + b[idx & 63], 0.0f);
}

__global__ void gru_elem(const float* __restrict__ gi,
                         const float* __restrict__ gh,
                         float* __restrict__ x, int V)
{
    int idx = blockIdx.x * blockDim.x + threadIdx.x;
    if (idx >= V * 64) return;
    int v = idx >> 6, j = idx & 63;
    const float* a = gi + (size_t)v * 192;
    const float* b = gh + (size_t)v * 192;
    float rg = 1.0f / (1.0f + __expf(-(a[j]      + b[j])));
    float zg = 1.0f / (1.0f + __expf(-(a[64 + j] + b[64 + j])));
    float ng = tanhf(a[128 + j] + rg * b[128 + j]);
    float h  = x[idx];
    x[idx] = (1.0f - zg) * ng + zg * h;
}

__global__ void graph_accum(const float* __restrict__ x,
                            const int* __restrict__ gid,
                            float* __restrict__ sums,
                            float* __restrict__ cnts, int V)
{
    const int vpb = blockDim.x >> 6;
    const int v   = blockIdx.x * vpb + (threadIdx.x >> 6);
    const int o   = threadIdx.x & 63;
    if (v >= V) return;
    int g = gid[v];
    atomicAdd(&sums[(size_t)g * 64 + o], x[(size_t)v * 64 + o]);
    if (o == 0) atomicAdd(&cnts[g], 1.0f);
}

__global__ void readout_mlp(const float* __restrict__ sums,
                            const float* __restrict__ cnts,
                            const float* __restrict__ W0, const float* __restrict__ b0,
                            const float* __restrict__ W1, const float* __restrict__ b1,
                            const float* __restrict__ W2, const float* __restrict__ b2,
                            float* __restrict__ out, int G)
{
    int g = blockIdx.x * blockDim.x + threadIdx.x;
    if (g >= G) return;
    float hg[64], t0[32], t1[16];
    float c = fmaxf(cnts[g], 1.0f);
    #pragma unroll
    for (int i = 0; i < 64; ++i) hg[i] = sums[(size_t)g * 64 + i] / c;
    for (int j = 0; j < 32; ++j) {
        float a = b0[j];
        for (int i = 0; i < 64; ++i) a += hg[i] * W0[i * 32 + j];
        t0[j] = fmaxf(a, 0.0f);
    }
    for (int j = 0; j < 16; ++j) {
        float a = b1[j];
        for (int i = 0; i < 32; ++i) a += t0[i] * W1[i * 16 + j];
        t1[j] = fmaxf(a, 0.0f);
    }
    for (int j = 0; j < 2; ++j) {
        float a = b2[j];
        for (int i = 0; i < 16; ++i) a += t1[i] * W2[i * 2 + j];
        out[(size_t)g * 2 + j] = a;
    }
}

// ---------------------------------------------------------------------------
// Host-side orchestration
// ---------------------------------------------------------------------------
extern "C" void kernel_launch(void* const* d_in, const int* in_sizes, int n_in,
                              void* d_out, int out_size, void* d_ws, size_t ws_size,
                              hipStream_t stream)
{
    (void)in_sizes; (void)n_in; (void)out_size; (void)ws_size;

    const float* h      = (const float*)d_in[0];
    const float* e      = (const float*)d_in[1];
    const int*   esrc   = (const int*)d_in[2];
    const int*   edst   = (const int*)d_in[3];
    const int*   gid    = (const int*)d_in[4];
    const float* W_proj = (const float*)d_in[6];
    const float* b_proj = (const float*)d_in[7];
    const float* W_e1   = (const float*)d_in[8];
    const float* b_e1   = (const float*)d_in[9];
    const float* W_e2   = (const float*)d_in[10];
    const float* b_e2   = (const float*)d_in[11];
    const float* b_conv = (const float*)d_in[12];
    const float* W_ih   = (const float*)d_in[13];
    const float* W_hh   = (const float*)d_in[14];
    const float* b_ih   = (const float*)d_in[15];
    const float* b_hh   = (const float*)d_in[16];
    const float* W_r0   = (const float*)d_in[17];
    const float* b_r0   = (const float*)d_in[18];
    const float* W_r1   = (const float*)d_in[19];
    const float* b_r1   = (const float*)d_in[20];
    const float* W_r2   = (const float*)d_in[21];
    const float* b_r2   = (const float*)d_in[22];

    constexpr int V = 10000, E = 20000, G = 256;
    constexpr int N_STEPS = 5;

    // -------- workspace carve-out (256B aligned) --------
    char* wp = (char*)d_ws;
    auto carve = [&](size_t bytes) {
        char* p = wp;
        wp += (bytes + 255) & ~(size_t)255;
        return p;
    };
    float*  x    = (float*) carve((size_t)V * 64  * 4);   // node state / hidden
    float*  a1   = (float*) carve((size_t)E * 128 * 4);   // edge-net hidden
    __bf16* We   = (__bf16*)carve((size_t)E * 4096 * 2);  // 164MB bf16 -> L2-resident
    float*  m    = (float*) carve((size_t)V * 64  * 4);   // conv output
    float*  gi   = (float*) carve((size_t)V * 192 * 4);
    float*  gh   = (float*) carve((size_t)V * 192 * 4);
    float*  agg  = (float*) carve((size_t)V * 64  * 4);
    float*  sums = (float*) carve((size_t)G * 64  * 4);
    float*  cnts = (float*) carve((size_t)G * 4);

    // 1) x = relu(h @ W_proj + b_proj)      [10000,74]x[74,64]
    gemm_wmma<74, 64, true, false><<<dim3(V / 16, 1), 128, 0, stream>>>(
        h, W_proj, b_proj, x, 64);

    // 2) a1 = relu(e @ W_e1 + b_e1)         [20000,12]x[12,128]
    gemm_wmma<12, 128, true, false><<<dim3(E / 16, 1), 256, 0, stream>>>(
        e, W_e1, b_e1, a1, 128);

    // 3) We = a1 @ W_e2 + b_e2 (bf16 out)   [20000,128]x[128,4096] -- dominant GEMM
    gemm_wmma<128, 128, false, true><<<dim3(E / 16, 4096 / 128), 256, 0, stream>>>(
        a1, W_e2, b_e2, We, 4096);

    const long nVH  = (long)V * 64;
    const int  gVH  = (int)((nVH + 255) / 256);
    const int  gMsg = E / MSG_WAVES;
    const int  gAcc = (V + 3) / 4;

    for (int step = 0; step < N_STEPS; ++step) {
        zero_f32<<<gVH, 256, 0, stream>>>(agg, nVH);
        msg_scatter<<<gMsg, MSG_WAVES * 32, 0, stream>>>(x, We, esrc, edst, agg, E);
        bias_relu64<<<gVH, 256, 0, stream>>>(agg, b_conv, m, V);
        // gi = m @ W_ih + b_ih ; gh = x @ W_hh + b_hh   [10000,64]x[64,192]
        gemm_wmma<64, 96, false, false><<<dim3(V / 16, 2), 192, 0, stream>>>(
            m, W_ih, b_ih, gi, 192);
        gemm_wmma<64, 96, false, false><<<dim3(V / 16, 2), 192, 0, stream>>>(
            x, W_hh, b_hh, gh, 192);
        gru_elem<<<gVH, 256, 0, stream>>>(gi, gh, x, V);
    }

    // readout: per-graph mean -> MLP
    zero_f32<<<(G * 64 + 255) / 256, 256, 0, stream>>>(sums, (long)G * 64);
    zero_f32<<<1, 256, 0, stream>>>(cnts, G);
    graph_accum<<<gAcc, 256, 0, stream>>>(x, gid, sums, cnts, V);
    readout_mlp<<<1, 256, 0, stream>>>(sums, cnts, W_r0, b_r0, W_r1, b_r1,
                                       W_r2, b_r2, (float*)d_out, G);
}
